// PhysicsAttention_12309376271061
// MI455X (gfx1250) — matrix operand order repaired and checked by reading, not compiled, not run on hardware
//
#include <hip/hip_runtime.h>
#include <hip/hip_bf16.h>

typedef __attribute__((ext_vector_type(16))) __bf16 v16bf;
typedef __attribute__((ext_vector_type(8)))  float  v8f;
typedef __attribute__((ext_vector_type(4)))  unsigned uv4;
typedef __attribute__((ext_vector_type(8)))  int  iv8;
typedef __attribute__((ext_vector_type(4)))  int  iv4;

#define DIMX 256
#define HEADS 8
#define DIM_HEAD 32
#define SLICE_NUM 64
#define BATCH 4
#define NTOK 32768
#define TILE_N 128

#if defined(__has_builtin)
#if __has_builtin(__builtin_amdgcn_tensor_load_to_lds) && \
    __has_builtin(__builtin_amdgcn_s_wait_tensorcnt)
#define HAVE_TDM 1
#endif
#endif

__device__ __forceinline__ unsigned short f2bf(float f) {
  unsigned u = __float_as_uint(f);
  u += 0x7FFFu + ((u >> 16) & 1u);   // round-to-nearest-even
  return (unsigned short)(u >> 16);
}

union FragBF { uint4 q[2]; v16bf v; };

// A fragment (16x32, M x K) from row-major bf16 [row][k], stride in halves.
__device__ __forceinline__ v16bf ldfragA(const unsigned short* base, int stride,
                                         int mtile, int k0, int lane) {
  int row = mtile * 16 + (lane & 15);
  int kb  = k0 + ((lane >> 4) << 4);
  const unsigned short* p = base + row * stride + kb;
  FragBF f;
  f.q[0] = *(const uint4*)(p);
  f.q[1] = *(const uint4*)(p + 8);
  return f.v;
}

// B fragment (32x16, K x N) from K-major bf16 [col][k] (transposed store).
__device__ __forceinline__ v16bf ldfragB_T(const unsigned short* base, int stride,
                                           int col0, int ktile, int lane) {
  int col = col0 + (lane & 15);
  int kb  = ktile * 32 + ((lane >> 4) << 4);
  const unsigned short* p = base + col * stride + kb;
  FragBF f;
  f.q[0] = *(const uint4*)(p);
  f.q[1] = *(const uint4*)(p + 8);
  return f.v;
}

// B fragment from pre-packed buffer: 16 contiguous bf16 per lane per tile.
__device__ __forceinline__ v16bf ldfragB_P(const unsigned short* packed, int lane) {
  const unsigned short* p = packed + lane * 16;
  FragBF f;
  f.q[0] = *(const uint4*)(p);
  f.q[1] = *(const uint4*)(p + 8);
  return f.v;
}

__device__ __forceinline__ v8f wmma_bf16(v16bf a, v16bf b, v8f c) {
  return __builtin_amdgcn_wmma_f32_16x16x32_bf16(false, a, false, b, (short)0, c,
                                                 false, false);
}

#ifdef HAVE_TDM
// One-shot TDM: DMA a [rows x cols] fp32 tile (row stride == cols) into LDS.
// Issued by a single wave; EXEC is ignored by TDM. D# per CDNA5 ISA ch.8.
// clang-23 toolchain: 6-arg builtin (g0 v4u, g1 v8i, g2 v4i, g3 v4i, v8i, cpol).
__device__ __forceinline__ void tdm_load_tile(const float* gsrc, unsigned ldsOff,
                                              int rows, int cols) {
  unsigned long long ga = (unsigned long long)(uintptr_t)gsrc;
  uv4 g0;
  g0[0] = 1u;                                           // count=1, user mode
  g0[1] = ldsOff;                                       // lds_addr (bytes)
  g0[2] = (unsigned)(ga & 0xFFFFFFFFu);                 // global_addr[31:0]
  g0[3] = (unsigned)((ga >> 32) & 0x1FFFFFFu) | (2u << 30);  // addr[56:32], type=2
  iv8 g1;
  g1[0] = 0x00020000;                                   // data_size=2 (4 bytes)
  g1[1] = (int)((unsigned)(cols & 0xFFFF) << 16);       // tensor_dim0[15:0]
  g1[2] = (int)(((unsigned)(cols >> 16) & 0xFFFFu) |
                ((unsigned)(rows & 0xFFFF) << 16));     // dim0 hi | tensor_dim1 lo
  g1[3] = (int)(((unsigned)(rows >> 16) & 0xFFFFu) |
                ((unsigned)(cols & 0xFFFF) << 16));     // dim1 hi | tile_dim0
  g1[4] = rows & 0xFFFF;                                // tile_dim1 (tile_dim2=0)
  g1[5] = cols;                                         // tensor_dim0_stride lo
  g1[6] = 0;
  g1[7] = 0;
  iv4 z4 = {0, 0, 0, 0};
  iv8 z8 = {0, 0, 0, 0, 0, 0, 0, 0};
  __builtin_amdgcn_tensor_load_to_lds(g0, g1, z4, z4, z8, 0);
}
#endif

// ---------------- weight pre-pack: row-major fp32 [K][N] -> B-fragment bf16 ----
__global__ void pack_w(const float* __restrict__ W, int K, int N,
                       unsigned short* __restrict__ dst) {
  int ktiles = K >> 5;
  int tile = blockIdx.x;              // tile = nt * ktiles + kt
  int nt = tile / ktiles, kt = tile - nt * ktiles;
  for (int i = threadIdx.x; i < 512; i += blockDim.x) {
    int lane = i >> 4, e = i & 15;
    int k = kt * 32 + ((lane >> 4) << 4) + e;
    int n = nt * 16 + (lane & 15);
    dst[(size_t)tile * 512 + i] = f2bf(W[(size_t)k * N + n]);
  }
}

__global__ void zero_ws(float* p, int n) {
  int i = blockIdx.x * blockDim.x + threadIdx.x;
  if (i < n) p[i] = 0.0f;
}

// ---------------- Kernel A: projections + gating softmax + pooling -------------
// LDS map (bytes):
//   xb   @ 0       [128][264] bf16   67584
//   fxT  @ 67584   [256][136] bf16   69632
//   xmid @ 137216  [128][264] bf16   67584
//   lg   @ 204800  [128][68]  f32    34816
//   wT   @ 239616  [64][136]  bf16   17408   -> total 257024
//   xstage @ 67584 [128][256] f32   131072 (overlays fxT/xmid; dead after cvt)
__global__ __launch_bounds__(256) void kA(
    const float* __restrict__ x, const float* __restrict__ b_fx,
    const float* __restrict__ b_x, const float* __restrict__ b_slice,
    const float* __restrict__ temperature,
    const unsigned short* __restrict__ pWfx, const unsigned short* __restrict__ pWx,
    const unsigned short* __restrict__ pWs,
    float* __restrict__ g_norm, float* __restrict__ g_token) {
  __shared__ __align__(16) char smem[257024];
  unsigned short* xb   = (unsigned short*)(smem);
  unsigned short* fxT  = (unsigned short*)(smem + 67584);
  unsigned short* xmid = (unsigned short*)(smem + 137216);
  float*          lg   = (float*)(smem + 204800);
  unsigned short* wT   = (unsigned short*)(smem + 239616);

  const int tid = threadIdx.x, lane = tid & 31, wave = tid >> 5;
  const int b  = blockIdx.x / (NTOK / TILE_N);
  const int n0 = (blockIdx.x % (NTOK / TILE_N)) * TILE_N;
  const float* xrow = x + ((size_t)b * NTOK + n0) * DIMX;

#ifdef HAVE_TDM
  float* xstage = (float*)(smem + 67584);
  if (wave == 0) {
    tdm_load_tile(xrow, (unsigned)(uintptr_t)xstage, TILE_N, DIMX);
    __builtin_amdgcn_s_wait_tensorcnt(0);
  }
  __syncthreads();
  for (int i = tid; i < TILE_N * DIMX / 4; i += 256) {
    int r = i >> 6, c4 = i & 63;
    float4 v = *(const float4*)(xstage + (size_t)i * 4);
    ushort4 h; h.x = f2bf(v.x); h.y = f2bf(v.y); h.z = f2bf(v.z); h.w = f2bf(v.w);
    *(ushort4*)&xb[r * 264 + c4 * 4] = h;
  }
#else
  for (int i = tid; i < TILE_N * DIMX / 4; i += 256) {
    int r = i >> 6, c4 = i & 63;
    float4 v = *(const float4*)(xrow + (size_t)r * DIMX + c4 * 4);
    ushort4 h; h.x = f2bf(v.x); h.y = f2bf(v.y); h.z = f2bf(v.z); h.w = f2bf(v.w);
    *(ushort4*)&xb[r * 264 + c4 * 4] = h;
  }
#endif
  __syncthreads();

  // projections: wave = mtile (8 tiles of 16 rows); 32 ntiles: 0-15 fx, 16-31 x_mid
  for (int nt = 0; nt < 32; ++nt) {
    v8f acc = {};
    const unsigned short* pw = (nt < 16)
        ? pWfx + (size_t)nt * 8 * 512
        : pWx  + (size_t)(nt - 16) * 8 * 512;
    for (int kt = 0; kt < 8; ++kt) {
      v16bf a  = ldfragA(xb, 264, wave, kt * 32, lane);
      v16bf bb = ldfragB_P(pw + kt * 512, lane);
      acc = wmma_bf16(a, bb, acc);
    }
    int colL = (nt & 15) * 16 + (lane & 15);
    if (nt < 16) {
      float bias = b_fx[colL];
      for (int r = 0; r < 8; ++r) {
        int row = wave * 16 + r + ((lane >> 4) << 3);
        fxT[colL * 136 + row] = f2bf(acc[r] + bias);
      }
    } else {
      float bias = b_x[colL];
      for (int r = 0; r < 8; ++r) {
        int row = wave * 16 + r + ((lane >> 4) << 3);
        xmid[row * 264 + colL] = f2bf(acc[r] + bias);
      }
    }
  }
  __syncthreads();

  for (int h = 0; h < HEADS; ++h) {
    // logits[n][g]: M=128 (wave=mtile), K=32, N=64 (4 ntiles)
    for (int nt = 0; nt < 4; ++nt) {
      v16bf a  = ldfragA(xmid, 264, wave, h * 32, lane);
      v16bf bb = ldfragB_P(pWs + (size_t)nt * 512, lane);
      v8f acc = {};
      acc = wmma_bf16(a, bb, acc);
      int colG = nt * 16 + (lane & 15);
      for (int r = 0; r < 8; ++r)
        lg[(wave * 16 + r + ((lane >> 4) << 3)) * 68 + colG] = acc[r];
    }
    __syncthreads();

    if (tid < TILE_N) {
      float inv_tau = 1.0f / temperature[h];
      float mx = -3.0e38f;
      for (int g = 0; g < 64; ++g) {
        float v = lg[tid * 68 + g] + b_slice[g];
        lg[tid * 68 + g] = v;
        mx = fmaxf(mx, v);
      }
      float s = 0.f;
      for (int g = 0; g < 64; ++g) {
        float e = __expf((lg[tid * 68 + g] - mx) * inv_tau);
        lg[tid * 68 + g] = e; s += e;
      }
      float inv_s = 1.0f / s;
      for (int g = 0; g < 64; ++g) {
        float w = lg[tid * 68 + g] * inv_s;
        lg[tid * 68 + g] = w;
        wT[g * 136 + tid] = f2bf(w);
      }
    }
    __syncthreads();

    if (tid < 64) {                       // slice_norm partial
      float s = 0.f;
      for (int r = 0; r < TILE_N; ++r) s += lg[r * 68 + tid];
      atomicAdd(&g_norm[((size_t)b * HEADS + h) * 64 + tid], s);
    }

    // pooling: token[g][c] += w^T @ fx ; 4 mtiles(G) x 2 ntiles(C) = 8 wave jobs
    {
      int mt = wave >> 1, ntc = wave & 1;
      v8f acc = {};
      for (int kt = 0; kt < 4; ++kt) {
        v16bf a  = ldfragA(wT, 136, mt, kt * 32, lane);
        v16bf bb = ldfragB_T(fxT, 136, h * 32 + ntc * 16, kt, lane);
        acc = wmma_bf16(a, bb, acc);
      }
      int cL = ntc * 16 + (lane & 15);
      float* tok = g_token + ((size_t)b * HEADS + h) * 64 * 32;
      for (int r = 0; r < 8; ++r) {
        int g = mt * 16 + r + ((lane >> 4) << 3);
        atomicAdd(&tok[g * 32 + cL], acc[r]);
      }
    }
    __syncthreads();
  }
}

// ---------------- Kernel B: tiny attention over G=64 slice tokens --------------
__global__ __launch_bounds__(128) void kB(
    const float* __restrict__ g_token, const float* __restrict__ g_norm,
    const float* __restrict__ Wq, const float* __restrict__ Wk,
    const float* __restrict__ Wv, unsigned short* __restrict__ osP) {
  __shared__ float tok[64][33];
  __shared__ float qs[64][33];
  __shared__ float ks[64][33];
  __shared__ float vs[64][33];
  __shared__ float at[64][65];
  __shared__ float os[64][33];
  int bh = blockIdx.x, tid = threadIdx.x;
  const float* tsrc = g_token + (size_t)bh * 64 * 32;
  const float* nsrc = g_norm + (size_t)bh * 64;

  for (int i = tid; i < 2048; i += 128) {
    int g = i >> 5, c = i & 31;
    tok[g][c] = tsrc[i] / (nsrc[g] + 1e-5f);
  }
  __syncthreads();
  for (int i = tid; i < 2048; i += 128) {
    int g = i >> 5, c = i & 31;
    float aq = 0, ak = 0, av = 0;
    for (int j = 0; j < 32; ++j) {
      float t = tok[g][j];
      aq += t * Wq[j * 32 + c];
      ak += t * Wk[j * 32 + c];
      av += t * Wv[j * 32 + c];
    }
    qs[g][c] = aq; ks[g][c] = ak; vs[g][c] = av;
  }
  __syncthreads();
  const float scale = 0.17677669529663687f;   // 1/sqrt(32)
  for (int i = tid; i < 4096; i += 128) {
    int g = i >> 6, k = i & 63;
    float s = 0;
    for (int c = 0; c < 32; ++c) s += qs[g][c] * ks[k][c];
    at[g][k] = s * scale;
  }
  __syncthreads();
  if (tid < 64) {
    float mx = -3.0e38f;
    for (int k = 0; k < 64; ++k) mx = fmaxf(mx, at[tid][k]);
    float s = 0;
    for (int k = 0; k < 64; ++k) { float e = __expf(at[tid][k] - mx); at[tid][k] = e; s += e; }
    float inv = 1.0f / s;
    for (int k = 0; k < 64; ++k) at[tid][k] *= inv;
  }
  __syncthreads();
  for (int i = tid; i < 2048; i += 128) {
    int g = i >> 5, c = i & 31;
    float s = 0;
    for (int k = 0; k < 64; ++k) s += at[g][k] * vs[k][c];
    os[g][c] = s;
  }
  __syncthreads();
  // emit out_slice pre-packed in B-fragment layout: [nt(2)][kt(2)][lane(32)][e(16)]
  unsigned short* dst = osP + (size_t)bh * 2048;
  for (int i = tid; i < 2048; i += 128) {
    int nt = i >> 10, kt = (i >> 9) & 1, lane = (i >> 4) & 31, e = i & 15;
    int g = kt * 32 + ((lane >> 4) << 4) + e;
    int c = nt * 16 + (lane & 15);
    dst[i] = f2bf(os[g][c]);
  }
}

// ---------------- Kernel C: recompute gating, unpool, output projection --------
// LDS map (bytes):
//   xb   @ 0       [128][264] bf16   67584
//   xmid @ 67584   [128][264] bf16   67584
//   lg   @ 135168  [128][68]  f32    34816
//   wrow @ 169984  [128][72]  bf16   18432
//   ox   @ 188416  [128][264] bf16   67584  -> total 256000
//   xstage @ 67584 [128][256] f32   131072 (overlays xmid/lg/wrow; dead after cvt)
__global__ __launch_bounds__(256) void kC(
    const float* __restrict__ x, const float* __restrict__ b_x,
    const float* __restrict__ b_slice, const float* __restrict__ temperature,
    const unsigned short* __restrict__ pWx, const unsigned short* __restrict__ pWs,
    const unsigned short* __restrict__ osP, const unsigned short* __restrict__ pWout,
    const float* __restrict__ b_out, float* __restrict__ out) {
  __shared__ __align__(16) char smem[256000];
  unsigned short* xb   = (unsigned short*)(smem);
  unsigned short* xmid = (unsigned short*)(smem + 67584);
  float*          lg   = (float*)(smem + 135168);
  unsigned short* wrow = (unsigned short*)(smem + 169984);
  unsigned short* ox   = (unsigned short*)(smem + 188416);

  const int tid = threadIdx.x, lane = tid & 31, wave = tid >> 5;
  const int b  = blockIdx.x / (NTOK / TILE_N);
  const int n0 = (blockIdx.x % (NTOK / TILE_N)) * TILE_N;
  const float* xrow = x + ((size_t)b * NTOK + n0) * DIMX;

#ifdef HAVE_TDM
  float* xstage = (float*)(smem + 67584);
  if (wave == 0) {
    tdm_load_tile(xrow, (unsigned)(uintptr_t)xstage, TILE_N, DIMX);
    __builtin_amdgcn_s_wait_tensorcnt(0);
  }
  __syncthreads();
  for (int i = tid; i < TILE_N * DIMX / 4; i += 256) {
    int r = i >> 6, c4 = i & 63;
    float4 v = *(const float4*)(xstage + (size_t)i * 4);
    ushort4 h; h.x = f2bf(v.x); h.y = f2bf(v.y); h.z = f2bf(v.z); h.w = f2bf(v.w);
    *(ushort4*)&xb[r * 264 + c4 * 4] = h;
  }
#else
  for (int i = tid; i < TILE_N * DIMX / 4; i += 256) {
    int r = i >> 6, c4 = i & 63;
    float4 v = *(const float4*)(xrow + (size_t)r * DIMX + c4 * 4);
    ushort4 h; h.x = f2bf(v.x); h.y = f2bf(v.y); h.z = f2bf(v.z); h.w = f2bf(v.w);
    *(ushort4*)&xb[r * 264 + c4 * 4] = h;
  }
#endif
  __syncthreads();

  for (int nt = 0; nt < 16; ++nt) {              // x_mid projection
    v8f acc = {};
    for (int kt = 0; kt < 8; ++kt) {
      v16bf a  = ldfragA(xb, 264, wave, kt * 32, lane);
      v16bf bb = ldfragB_P(pWx + ((size_t)nt * 8 + kt) * 512, lane);
      acc = wmma_bf16(a, bb, acc);
    }
    int colL = nt * 16 + (lane & 15);
    float bias = b_x[colL];
    for (int r = 0; r < 8; ++r)
      xmid[(wave * 16 + r + ((lane >> 4) << 3)) * 264 + colL] = f2bf(acc[r] + bias);
  }
  __syncthreads();

  for (int h = 0; h < HEADS; ++h) {
    for (int nt = 0; nt < 4; ++nt) {
      v16bf a  = ldfragA(xmid, 264, wave, h * 32, lane);
      v16bf bb = ldfragB_P(pWs + (size_t)nt * 512, lane);
      v8f acc = {};
      acc = wmma_bf16(a, bb, acc);
      int colG = nt * 16 + (lane & 15);
      for (int r = 0; r < 8; ++r)
        lg[(wave * 16 + r + ((lane >> 4) << 3)) * 68 + colG] = acc[r];
    }
    __syncthreads();

    if (tid < TILE_N) {
      float inv_tau = 1.0f / temperature[h];
      float mx = -3.0e38f;
      for (int g = 0; g < 64; ++g) {
        float v = lg[tid * 68 + g] + b_slice[g];
        lg[tid * 68 + g] = v;
        mx = fmaxf(mx, v);
      }
      float s = 0.f;
      for (int g = 0; g < 64; ++g) {
        float e = __expf((lg[tid * 68 + g] - mx) * inv_tau);
        lg[tid * 68 + g] = e; s += e;
      }
      float inv_s = 1.0f / s;
      for (int g = 0; g < 64; ++g)
        wrow[tid * 72 + g] = f2bf(lg[tid * 68 + g] * inv_s);
    }
    __syncthreads();

    // unpool: out_x[n][c] = w @ out_slice ; wave = mtile, 2 C-ntiles each
    for (int ntc = 0; ntc < 2; ++ntc) {
      v8f acc = {};
      for (int kt = 0; kt < 2; ++kt) {
        v16bf a = ldfragA(wrow, 72, wave, kt * 32, lane);
        const unsigned short* p =
            osP + ((((size_t)b * HEADS + h) * 2 + ntc) * 2 + kt) * 512;
        v16bf bb = ldfragB_P(p, lane);
        acc = wmma_bf16(a, bb, acc);
      }
      int colL = h * 32 + ntc * 16 + (lane & 15);
      for (int r = 0; r < 8; ++r)
        ox[(wave * 16 + r + ((lane >> 4) << 3)) * 264 + colL] = f2bf(acc[r]);
    }
    __syncthreads();
  }

  // final projection: out = ox @ W_out + b_out
  float* orow = out + ((size_t)b * NTOK + n0) * DIMX;
  for (int nt = 0; nt < 16; ++nt) {
    v8f acc = {};
    for (int kt = 0; kt < 8; ++kt) {
      v16bf a  = ldfragA(ox, 264, wave, kt * 32, lane);
      v16bf bb = ldfragB_P(pWout + ((size_t)nt * 8 + kt) * 512, lane);
      acc = wmma_bf16(a, bb, acc);
    }
    int col = nt * 16 + (lane & 15);
    float bias = b_out[col];
    for (int r = 0; r < 8; ++r) {
      int row = wave * 16 + r + ((lane >> 4) << 3);
      orow[(size_t)row * DIMX + col] = acc[r] + bias;
    }
  }
}

extern "C" void kernel_launch(void* const* d_in, const int* in_sizes, int n_in,
                              void* d_out, int out_size, void* d_ws, size_t ws_size,
                              hipStream_t stream) {
  (void)in_sizes; (void)n_in; (void)out_size; (void)ws_size;
  const float* x     = (const float*)d_in[0];
  const float* W_fx  = (const float*)d_in[1];
  const float* b_fx  = (const float*)d_in[2];
  const float* W_x   = (const float*)d_in[3];
  const float* b_x   = (const float*)d_in[4];
  const float* W_sl  = (const float*)d_in[5];
  const float* b_sl  = (const float*)d_in[6];
  const float* temp  = (const float*)d_in[7];
  const float* W_q   = (const float*)d_in[8];
  const float* W_k   = (const float*)d_in[9];
  const float* W_v   = (const float*)d_in[10];
  const float* W_out = (const float*)d_in[11];
  const float* b_out = (const float*)d_in[12];
  float* out = (float*)d_out;

  char* ws = (char*)d_ws;
  float*          g_norm  = (float*)(ws + 0);                 //   8192 B
  float*          g_token = (float*)(ws + 8192);              // 262144 B
  unsigned short* osP     = (unsigned short*)(ws + 270336);   // 131072 B
  unsigned short* pWfx    = (unsigned short*)(ws + 401408);   // 131072 B
  unsigned short* pWx     = (unsigned short*)(ws + 532480);   // 131072 B
  unsigned short* pWs     = (unsigned short*)(ws + 663552);   //   4096 B
  unsigned short* pWout   = (unsigned short*)(ws + 667648);   // 131072 B

  zero_ws<<<264, 256, 0, stream>>>(g_norm, 2048 + 65536);
  pack_w<<<128, 256, 0, stream>>>(W_fx, 256, 256, pWfx);
  pack_w<<<128, 256, 0, stream>>>(W_x,  256, 256, pWx);
  pack_w<<<4,   256, 0, stream>>>(W_sl,  32,  64, pWs);
  pack_w<<<128, 256, 0, stream>>>(W_out, 256, 256, pWout);

  kA<<<BATCH * (NTOK / TILE_N), 256, 0, stream>>>(
      x, b_fx, b_x, b_sl, temp, pWfx, pWx, pWs, g_norm, g_token);
  kB<<<BATCH * HEADS, 128, 0, stream>>>(g_token, g_norm, W_q, W_k, W_v, osP);
  kC<<<BATCH * (NTOK / TILE_N), 256, 0, stream>>>(
      x, b_x, b_sl, temp, pWx, pWs, osP, pWout, b_out, out);
}
